// CCE_67190468378878
// MI455X (gfx1250) — matrix operand order repaired and checked by reading, not compiled, not run on hardware
//
#include <hip/hip_runtime.h>

// ---- problem constants (from reference setup) ----
#define B_N 2048
#define C_N 200
#define P_N 32
#define F_N 768
#define FP  776              // padded LDS row (avoid systematic bank aliasing: 768 f32-banks stride)
#define NWAVES 4
#define ALPHA_C 5.0f
#define BETA_C  5.0f

typedef __attribute__((ext_vector_type(16))) _Float16 v16h;
typedef __attribute__((ext_vector_type(8)))  float    v8f;

// -------------------------------------------------------------------------
// Kernel 1: deterministic bucketing of samples by target class.
// Single block, 1024 threads. Rank within class = #{b' < b : tc[b']==tc[b]}
// computed from an LDS copy of tc -> bucket order is deterministic.
// -------------------------------------------------------------------------
__global__ void cce_bucket_kernel(const int* __restrict__ tc,
                                  int* __restrict__ offsets,   // [C_N+1]
                                  int* __restrict__ bucket) {  // [B_N]
    __shared__ int tcs[B_N];
    __shared__ int cnt[C_N];
    __shared__ int offsL[C_N];
    const int t = threadIdx.x;

    for (int c = t; c < C_N; c += 1024) cnt[c] = 0;
    __syncthreads();
    for (int b = t; b < B_N; b += 1024) {
        int c = tc[b];
        tcs[b] = c;
        atomicAdd(&cnt[c], 1);           // counts are order-independent
    }
    __syncthreads();
    if (t == 0) {
        int s = 0;
        for (int c = 0; c < C_N; ++c) { offsL[c] = s; offsets[c] = s; s += cnt[c]; }
        offsets[C_N] = s;
    }
    __syncthreads();
    for (int b = t; b < B_N; b += 1024) {
        int c = tcs[b];
        int r = 0;
        for (int bp = 0; bp < b; ++bp) r += (tcs[bp] == c) ? 1 : 0;   // stable rank
        bucket[offsL[c] + r] = b;
    }
}

// -------------------------------------------------------------------------
// Kernel 2: one block per class. WMMA distance GEMM + exact fp32 MSE.
// 128 threads = 4 waves. Prototypes staged to LDS as f16.
// -------------------------------------------------------------------------
__global__ void __launch_bounds__(128)
cce_class_kernel(const float* __restrict__ outputs,   // [B_N, F_N]
                 const float* __restrict__ clusters,  // [C_N, P_N, F_N]
                 const int* __restrict__ offsets,     // [C_N+1]
                 const int* __restrict__ bucket,      // [B_N]
                 float* __restrict__ blockPart) {     // [C_N]
    __shared__ _Float16 protoH[P_N][FP];      // ~48.5 KB
    __shared__ float    c2s[P_N];
    __shared__ float    part[4][P_N];
    __shared__ int      pminTile[NWAVES][16];
    __shared__ float    wavePart[NWAVES];

    const int c    = blockIdx.x;
    const int tid  = threadIdx.x;
    const int wave = tid >> 5;
    const int lane = tid & 31;

    const float* cbase = clusters + (size_t)c * (P_N * F_N);

    // ---- stage prototypes (coalesced) as f16 into LDS ----
    for (int idx = tid; idx < P_N * F_N; idx += 128) {
        int p = idx / F_N, f = idx - p * F_N;
        protoH[p][f] = (_Float16)cbase[idx];
    }
    __syncthreads();

    // ---- deterministic c2[p] = sum_f proto^2 (fp32, from the f16 values used by WMMA) ----
    {
        int p = tid & 31, q = tid >> 5;                 // q in 0..3, 192 elems each
        float s = 0.f;
        for (int f = q * 192; f < q * 192 + 192; ++f) {
            float v = (float)protoH[p][f];
            s += v * v;
        }
        part[q][p] = s;
    }
    __syncthreads();
    if (tid < P_N) c2s[tid] = ((part[0][tid] + part[1][tid]) + part[2][tid]) + part[3][tid];
    __syncthreads();

    const int start = offsets[c];
    const int n     = offsets[c + 1] - start;
    const int nTiles = (n + 15) >> 4;

    float wsum = 0.f;

    for (int tile = wave; tile < nTiles; tile += NWAVES) {
        const int r0 = tile << 4;
        const int m  = lane & 15;
        const int gm = r0 + m;
        const int sIdx = (gm < n) ? bucket[start + gm] : -1;
        const float* xrow = outputs + (size_t)(sIdx < 0 ? 0 : sIdx) * F_N;

        v8f acc0 = {}, acc1 = {};

        // K loop: 768 in steps of 32 (f16 WMMA)
        for (int k0 = 0; k0 < F_N; k0 += 32) {
            // A fragment: 16-bit A 16x32 layout.
            // lanes 0-15 (M=lane):  elems 0..7 = K k0+0..7,  elems 8..15 = K k0+16..23
            // lanes 16-31 (M=lane-16): elems 0..7 = K k0+8..15, elems 8..15 = K k0+24..31
            v16h a = {};
            if (sIdx >= 0) {
                const int kbase = k0 + ((lane < 16) ? 0 : 8);
#pragma unroll
                for (int i = 0; i < 8; ++i) a[i]     = (_Float16)xrow[kbase + i];
#pragma unroll
                for (int i = 0; i < 8; ++i) a[8 + i] = (_Float16)xrow[kbase + 16 + i];
            }
            // B fragments: 16-bit B 32x16 layout.
            // lanes 0-15: N=lane,   elems 0..7 = K k0+0..7,  elems 8..15 = K k0+16..23
            // lanes 16-31: N=lane-16, elems 0..7 = K k0+8..15, elems 8..15 = K k0+24..31
            const int kb = k0 + ((lane < 16) ? 0 : 8);
            const int p0 = lane & 15;
            v16h b0, b1;
#pragma unroll
            for (int i = 0; i < 16; ++i) {
                b0[i] = protoH[p0][kb + i + (i >= 8 ? 8 : 0)];
                b1[i] = protoH[p0 + 16][kb + i + (i >= 8 ? 8 : 0)];
            }
            acc0 = __builtin_amdgcn_wmma_f32_16x16x32_f16(false, a, false, b0,
                                                          (short)0, acc0, false, false);
            acc1 = __builtin_amdgcn_wmma_f32_16x16x32_f16(false, a, false, b1,
                                                          (short)0, acc1, false, false);
        }

        // ---- per-row argmin over 32 prototypes: argmin_p (c2[p] - 2*xc[b,p]) ----
        // D layout: elem r -> lanes 0-15: (M=r, N=lane); lanes 16-31: (M=r+8, N=lane-16)
        const int pA = lane & 15;   // acc0 column
        const int pB = pA + 16;     // acc1 column
#pragma unroll
        for (int r = 0; r < 8; ++r) {
            float vA = c2s[pA] - 2.0f * acc0[r];
            float vB = c2s[pB] - 2.0f * acc1[r];
            float v; int p;
            if (vB < vA) { v = vB; p = pB; } else { v = vA; p = pA; }
#pragma unroll
            for (int mask = 1; mask <= 8; mask <<= 1) {   // butterfly within 16-lane half
                float ov = __shfl_xor(v, mask, 32);
                int   op = __shfl_xor(p, mask, 32);
                if (ov < v || (ov == v && op < p)) { v = ov; p = op; }
            }
            if (lane == 0)       pminTile[wave][r]     = p;  // row r
            else if (lane == 16) pminTile[wave][r + 8] = p;  // row r+8
        }
        asm volatile("s_wait_dscnt 0" ::: "memory");  // same-wave LDS handoff

        // ---- exact fp32 MSE vs chosen prototype: lanes (m, m+16) split row m's 768 ----
        {
            float ps = 0.f;
            if (gm < n) {
                const int pm = pminTile[wave][m];
                const float* xr = outputs + (size_t)sIdx * F_N;
                const float* cr = cbase + (size_t)pm * F_N;
                const int f0 = (lane >> 4) * 384;
                for (int f = f0; f < f0 + 384; ++f) {
                    float d = xr[f] - cr[f];
                    ps += d * d;
                }
            }
            ps += __shfl_xor(ps, 16, 32);   // combine the two halves of the row
            if (lane < 16) wsum += ps;       // count each row once
        }
    }

    // ---- deterministic in-wave + in-block reduction ----
#pragma unroll
    for (int mask = 16; mask >= 1; mask >>= 1) wsum += __shfl_xor(wsum, mask, 32);
    if (lane == 0) wavePart[wave] = wsum;
    __syncthreads();
    if (tid == 0) {
        float s = ((wavePart[0] + wavePart[1]) + wavePart[2]) + wavePart[3];
        blockPart[c] = s;
    }
}

// -------------------------------------------------------------------------
// Kernel 3: finalize. t = sum/(B*F); out = ALPHA*t + BETA*(1-t)  (== 5.0).
// -------------------------------------------------------------------------
__global__ void cce_finalize_kernel(const float* __restrict__ blockPart,
                                    float* __restrict__ out) {
    if (threadIdx.x == 0 && blockIdx.x == 0) {
        float s = 0.f;
        for (int i = 0; i < C_N; ++i) s += blockPart[i];
        float t = s / (float)(B_N * F_N);
        out[0] = ALPHA_C * t + BETA_C * (1.0f - t);
    }
}

extern "C" void kernel_launch(void* const* d_in, const int* in_sizes, int n_in,
                              void* d_out, int out_size, void* d_ws, size_t ws_size,
                              hipStream_t stream) {
    const float* outputs  = (const float*)d_in[0];   // [2048, 768] f32
    const float* clusters = (const float*)d_in[1];   // [200, 32, 768] f32
    const int*   tclasses = (const int*)d_in[2];     // [2048] i32
    float* out = (float*)d_out;                       // scalar f32

    // workspace layout (all fully written by kernels before being read)
    float* blockPart = (float*)d_ws;                  // 200 floats
    int*   offsets   = (int*)d_ws + 256;              // 201 ints
    int*   bucket    = (int*)d_ws + 512;              // 2048 ints

    cce_bucket_kernel<<<1, 1024, 0, stream>>>(tclasses, offsets, bucket);
    cce_class_kernel<<<C_N, 128, 0, stream>>>(outputs, clusters, offsets, bucket, blockPart);
    cce_finalize_kernel<<<1, 32, 0, stream>>>(blockPart, out);
}
